// SelfAttention2d_63840393888280
// MI455X (gfx1250) — matrix operand order repaired and checked by reading, not compiled, not run on hardware
//
#include <hip/hip_runtime.h>
#include <hip/hip_bf16.h>

// ---------------------------------------------------------------------------
// SelfAttention2d for MI455X (gfx1250): bf16 WMMA everywhere, fused
// flash-attention (no NxN materialization). TDM / async-LDS tile staging.
//   B=16, C=512, H=W=48 -> N=2304, INTER=64
// ---------------------------------------------------------------------------

#define B_   16
#define C_   512
#define N_   2304
#define I_   64

typedef __attribute__((ext_vector_type(16))) __bf16 v16bf;
typedef __attribute__((ext_vector_type(8)))  float  v8f;
typedef unsigned int u32x4 __attribute__((ext_vector_type(4)));
typedef int          i32x4 __attribute__((ext_vector_type(4)));
typedef int          i32x8 __attribute__((ext_vector_type(8)));

// --- gfx1250 feature guards (host pass + unknown-toolchain safe) -----------
#if defined(__gfx1250__) && __has_builtin(__builtin_amdgcn_tensor_load_to_lds) && \
    __has_builtin(__builtin_amdgcn_s_wait_tensorcnt)
#define USE_TDM 1
#endif
#if defined(__gfx1250__)
#define USE_ASYNC 1
#endif

__device__ __forceinline__ v8f wmma_bf16(v16bf a, v16bf b, v8f c) {
  // 8 args: (neg_a, A, neg_b, B, c_mod, C, reuse_a, reuse_b)
  return __builtin_amdgcn_wmma_f32_16x16x32_bf16(false, a, false, b,
                                                 (short)0, c, false, false);
}

// Assemble a 32-byte WMMA operand from two 16-byte chunks (LDS or global).
__device__ __forceinline__ v16bf ldfrag(const __bf16* p0, const __bf16* p1) {
  union { v16bf v; float4 f[2]; } u;
  u.f[0] = *reinterpret_cast<const float4*>(p0);
  u.f[1] = *reinterpret_cast<const float4*>(p1);
  return u.v;
}

#if defined(USE_TDM)
// TDM 2D tile load: tile_k bf16 elements per row x tile_rows rows, row stride
// row_stride_elems, global -> LDS. D# per CDNA5 ISA ch.8 (data_size=2B,
// type=2, count=1). Issued once per wave; EXEC ignored by TDM.
__device__ __forceinline__ void tdm_load_tile_bf16(const void* gsrc, void* ldst,
                                                   unsigned tile_k,
                                                   unsigned tile_rows,
                                                   unsigned row_stride_elems) {
  const unsigned long long ga = (unsigned long long)(size_t)gsrc;
  const unsigned la = (unsigned)(size_t)ldst;   // generic LDS addr low bits
  u32x4 g0;
  g0[0] = 1u;                                            // count=1
  g0[1] = la;                                            // lds_addr
  g0[2] = (unsigned)ga;                                  // global_addr lo
  g0[3] = (unsigned)((ga >> 32) & 0x01FFFFFFu) | (2u << 30);  // addr hi | type=2
  i32x8 g1;
  g1[0] = (int)(1u << 16);                               // data_size=1 -> 2B
  g1[1] = (int)((row_stride_elems & 0xFFFFu) << 16);     // tensor_dim0 lo16
  g1[2] = (int)(row_stride_elems >> 16);                 // dim0 hi | dim1 lo=0
  g1[3] = (int)(16u | (tile_k << 16));                   // dim1 hi | tile_dim0
  g1[4] = (int)(tile_rows & 0xFFFFu);                    // tile_dim1 | tile_dim2=0
  g1[5] = (int)row_stride_elems;                         // tensor_dim0_stride lo
  g1[6] = 0;
  g1[7] = 0;
  i32x4 gz4 = {};
#if __clang_major__ >= 23
  i32x8 gz8 = {};
  __builtin_amdgcn_tensor_load_to_lds(g0, g1, gz4, gz4, gz8, 0);
#else
  __builtin_amdgcn_tensor_load_to_lds(g0, g1, gz4, gz4, 0);
#endif
}
#endif

#if defined(USE_ASYNC)
// ASYNCcnt-tracked global -> LDS copy (VGLOBAL encoding, GV mode):
//   global_load_async_to_lds_b128 vdst(LDS addr), vaddr(64b global), off
__device__ __forceinline__ void async_b128(const void* gsrc, void* ldst) {
  asm volatile("global_load_async_to_lds_b128 %0, %1, off"
               :
               : "v"((unsigned)(size_t)ldst),
                 "v"((unsigned long long)(size_t)gsrc)
               : "memory");
}
__device__ __forceinline__ void wait_asynccnt0() {
  asm volatile("s_wait_asynccnt 0x0" ::: "memory");
}
#endif

// ---------------------------------------------------------------------------
// Stage 0a: x (B,C,N) f32  ->  xT (B,N,C) bf16   (32x32 LDS tile transpose)
// ---------------------------------------------------------------------------
__global__ void prep_x(const float* __restrict__ x, __bf16* __restrict__ xT) {
  __shared__ float tile[32][33];
  const int b  = blockIdx.z;
  const int n0 = blockIdx.x * 32;
  const int c0 = blockIdx.y * 32;
  const int tx = threadIdx.x, ty = threadIdx.y;   // 32 x 8
  const float* xb = x + (size_t)b * C_ * N_;
  #pragma unroll
  for (int i = 0; i < 32; i += 8)
    tile[ty + i][tx] = xb[(size_t)(c0 + ty + i) * N_ + n0 + tx];
  __syncthreads();
  __bf16* o = xT + (size_t)b * N_ * C_;
  #pragma unroll
  for (int i = 0; i < 32; i += 8)
    o[(size_t)(n0 + ty + i) * C_ + c0 + tx] = (__bf16)tile[tx][ty + i];
}

// ---------------------------------------------------------------------------
// Stage 0b: weights -> bf16.  WqkT (512 x 128): cols 0..63 = Wq^T, 64..127 = Wk^T.
// ---------------------------------------------------------------------------
__global__ void prep_w(const float* __restrict__ Wq, const float* __restrict__ Wk,
                       const float* __restrict__ Wv,
                       __bf16* __restrict__ WqkT, __bf16* __restrict__ Wvb) {
  const int idx = blockIdx.x * 256 + threadIdx.x;     // grid covers 512*512
  if (idx < C_ * 128) {
    const int k = idx >> 7, c = idx & 127;
    const float v = (c < I_) ? Wq[c * C_ + k] : Wk[(c - I_) * C_ + k];
    WqkT[idx] = (__bf16)v;
  }
  if (idx < C_ * C_) Wvb[idx] = (__bf16)Wv[idx];
}

// ---------------------------------------------------------------------------
// Stage 1: QKT (B, N, 128) = xT (N x 512) * WqkT (512 x 128), per batch.
// 128x128 tile / workgroup, 8 waves (4x2), each wave 32x64.
// A-tile staged by the Tensor Data Mover when available.
// ---------------------------------------------------------------------------
__global__ void qk_gemm(const __bf16* __restrict__ xT,
                        const __bf16* __restrict__ WqkT,
                        __bf16* __restrict__ QKT) {
  __shared__ __attribute__((aligned(16))) __bf16 at[128 * 32];  // [row][k]
  __shared__ __attribute__((aligned(16))) __bf16 bt[128 * 32];  // [col][k]
  const int b   = blockIdx.z;
  const int n0  = blockIdx.x * 128;
  const int tid = threadIdx.x;
  const int lane = tid & 31, wave = tid >> 5;
  const int wrow = wave & 3, wcol = wave >> 2;
  const int lg = lane >> 4, li = lane & 15;
  const __bf16* A = xT + (size_t)b * N_ * C_;

  v8f acc[2][4];
  #pragma unroll
  for (int s = 0; s < 2; ++s)
    #pragma unroll
    for (int t = 0; t < 4; ++t) acc[s][t] = (v8f){};

  for (int kc = 0; kc < C_; kc += 32) {
#if defined(USE_TDM)
    if (wave == 0) {  // TDM: 128 rows x 32 bf16 from xT into at[]
      tdm_load_tile_bf16(A + (size_t)n0 * C_ + kc, at, 32u, 128u, (unsigned)C_);
    }
#else
    {  // stage A tile (128 rows x 32 k)
      const int row = tid >> 1, half = tid & 1;
      const __bf16* src = A + (size_t)(n0 + row) * C_ + kc + half * 16;
      *(float4*)&at[row * 32 + half * 16]     = *(const float4*)(src);
      *(float4*)&at[row * 32 + half * 16 + 8] = *(const float4*)(src + 8);
    }
#endif
    {  // stage B tile transposed: bt[col][k]
      const int col = tid & 127, kq = (tid >> 7) * 16;
      alignas(16) __bf16 tmp[16];
      #pragma unroll
      for (int j = 0; j < 16; ++j) tmp[j] = WqkT[(kc + kq + j) * 128 + col];
      *(float4*)&bt[col * 32 + kq]     = *(float4*)(tmp);
      *(float4*)&bt[col * 32 + kq + 8] = *(float4*)(tmp + 8);
    }
#if defined(USE_TDM)
    if (wave == 0) __builtin_amdgcn_s_wait_tensorcnt(0);
#endif
    __syncthreads();

    v16bf af[2], bfr[4];
    #pragma unroll
    for (int s = 0; s < 2; ++s) {
      const __bf16* p = &at[(wrow * 32 + s * 16 + li) * 32 + lg * 16];
      af[s] = ldfrag(p, p + 8);                   // K = lg*16 .. +15
    }
    #pragma unroll
    for (int t = 0; t < 4; ++t) {
      const __bf16* p = &bt[(wcol * 64 + t * 16 + li) * 32 + lg * 16];
      bfr[t] = ldfrag(p, p + 8);                  // K = lg*16 .. +15
    }
    #pragma unroll
    for (int s = 0; s < 2; ++s)
      #pragma unroll
      for (int t = 0; t < 4; ++t) acc[s][t] = wmma_bf16(af[s], bfr[t], acc[s][t]);
    __syncthreads();
  }

  __bf16* out = QKT + (size_t)b * N_ * 128;
  #pragma unroll
  for (int s = 0; s < 2; ++s)
    #pragma unroll
    for (int t = 0; t < 4; ++t)
      #pragma unroll
      for (int r = 0; r < 8; ++r) {
        const int row = n0 + wrow * 32 + s * 16 + r + 8 * lg;   // C-layout rows
        const int col = wcol * 64 + t * 16 + li;
        out[(size_t)row * 128 + col] = (__bf16)acc[s][t][r];
      }
}

// ---------------------------------------------------------------------------
// Stage 2: V (B, C, M) bf16 = Wv (512x512) * x (512 x 2304), per batch.
// A-tile (pure copy) staged with ASYNCcnt-tracked global->LDS b128 loads.
// ---------------------------------------------------------------------------
__global__ void v_gemm(const float* __restrict__ x,
                       const __bf16* __restrict__ Wvb,
                       __bf16* __restrict__ V) {
  __shared__ __attribute__((aligned(16))) __bf16 at[128 * 32];  // Wv rows
  __shared__ __attribute__((aligned(16))) __bf16 bt[128 * 32];  // [m][k]
  const int b   = blockIdx.z;
  const int m0  = blockIdx.x * 128;
  const int c0  = blockIdx.y * 128;
  const int tid = threadIdx.x;
  const int lane = tid & 31, wave = tid >> 5;
  const int wrow = wave & 3, wcol = wave >> 2;
  const int lg = lane >> 4, li = lane & 15;

  v8f acc[2][4];
  #pragma unroll
  for (int s = 0; s < 2; ++s)
    #pragma unroll
    for (int t = 0; t < 4; ++t) acc[s][t] = (v8f){};

  for (int kc = 0; kc < C_; kc += 32) {
    {  // A tile: Wv rows c0..c0+127, k chunk
      const int row = tid >> 1, half = tid & 1;
      const __bf16* src = Wvb + (size_t)(c0 + row) * C_ + kc + half * 16;
      __bf16* dst = &at[row * 32 + half * 16];
#if defined(USE_ASYNC)
      async_b128(src, dst);
      async_b128(src + 8, dst + 8);
#else
      *(float4*)(dst)     = *(const float4*)(src);
      *(float4*)(dst + 8) = *(const float4*)(src + 8);
#endif
    }
    {  // B tile: x[b, kc+kk, m0+mm] f32 -> bt[mm][kk] bf16
      const int kk = tid >> 3, moff = (tid & 7) * 16;
      const float* src = x + ((size_t)b * C_ + kc + kk) * N_ + m0 + moff;
      alignas(16) float f[16];
      *(float4*)&f[0]  = *(const float4*)(src);
      *(float4*)&f[4]  = *(const float4*)(src + 4);
      *(float4*)&f[8]  = *(const float4*)(src + 8);
      *(float4*)&f[12] = *(const float4*)(src + 12);
      #pragma unroll
      for (int j = 0; j < 16; ++j) bt[(moff + j) * 32 + kk] = (__bf16)f[j];
    }
#if defined(USE_ASYNC)
    wait_asynccnt0();
#endif
    __syncthreads();

    v16bf af[2], bfr[4];
    #pragma unroll
    for (int s = 0; s < 2; ++s) {
      const __bf16* p = &at[(wrow * 32 + s * 16 + li) * 32 + lg * 16];
      af[s] = ldfrag(p, p + 8);
    }
    #pragma unroll
    for (int t = 0; t < 4; ++t) {
      const __bf16* p = &bt[(wcol * 64 + t * 16 + li) * 32 + lg * 16];
      bfr[t] = ldfrag(p, p + 8);
    }
    #pragma unroll
    for (int s = 0; s < 2; ++s)
      #pragma unroll
      for (int t = 0; t < 4; ++t) acc[s][t] = wmma_bf16(af[s], bfr[t], acc[s][t]);
    __syncthreads();
  }

  #pragma unroll
  for (int s = 0; s < 2; ++s)
    #pragma unroll
    for (int t = 0; t < 4; ++t)
      #pragma unroll
      for (int r = 0; r < 8; ++r) {
        const int c = c0 + wrow * 32 + s * 16 + r + 8 * lg;
        const int m = m0 + wcol * 64 + t * 16 + li;
        V[((size_t)b * C_ + c) * N_ + m] = (__bf16)acc[s][t][r];
      }
}

// ---------------------------------------------------------------------------
// Stage 3: fused flash-attention + epilogue.
// One 16-row query tile per workgroup; 8 waves, wave w owns output channels
// [64w, 64w+64). Wave 0 computes S(16x32) + online softmax, shares P via LDS.
// ---------------------------------------------------------------------------
__global__ void attn_fused(const __bf16* __restrict__ QKT,
                           const __bf16* __restrict__ V,
                           const float* __restrict__ x,
                           const float* __restrict__ gamma,
                           float* __restrict__ out) {
  __shared__ __attribute__((aligned(16))) __bf16 pLds[16 * 32];  // P tile
  __shared__ float scaleLds[16];
  __shared__ float rsumLds[16];

  const int b   = blockIdx.y;
  const int n0  = blockIdx.x * 16;
  const int tid = threadIdx.x;
  const int lane = tid & 31, wave = tid >> 5;
  const int lg = lane >> 4, li = lane & 15;
  const int cw = wave * 64;

  const __bf16* QK = QKT + (size_t)b * N_ * 128;
  const __bf16* Vb = V   + (size_t)b * C_ * N_;

  // Q fragments for rows n0..n0+15 (A operand, K=64 in two 32-chunks)
  v16bf qf[2];
  #pragma unroll
  for (int kc2 = 0; kc2 < 2; ++kc2) {
    const __bf16* p = QK + (size_t)(n0 + li) * 128 + kc2 * 32 + lg * 16;
    qf[kc2] = ldfrag(p, p + 8);
  }

  v8f o[4];
  #pragma unroll
  for (int t = 0; t < 4; ++t) o[t] = (v8f){};
  float runmax[8], runsum[8];
  #pragma unroll
  for (int r = 0; r < 8; ++r) { runmax[r] = -3.0e38f; runsum[r] = 0.0f; }

  for (int m0 = 0; m0 < N_; m0 += 32) {
    // prefetch next V tile rows for this wave (global_prefetch path)
    if (m0 + 32 < N_) {
      const __bf16* pfp = Vb + (size_t)(cw + li) * N_ + (m0 + 32) + lg * 16;
      __builtin_prefetch(pfp, 0, 1);
    }
    if (wave == 0) {
      // S(16x32) = Q(16x64) . K^T : 2 col-tiles x 2 K-chunks = 4 WMMA
      v8f s[2] = { (v8f){}, (v8f){} };
      #pragma unroll
      for (int t = 0; t < 2; ++t)
        #pragma unroll
        for (int kc2 = 0; kc2 < 2; ++kc2) {
          const __bf16* p = QK + (size_t)(m0 + t * 16 + li) * 128
                               + I_ + kc2 * 32 + lg * 16;   // K block at col 64
          v16bf kf = ldfrag(p, p + 8);
          s[t] = wmma_bf16(qf[kc2], kf, s[t]);
        }
      // online softmax over this 32-wide tile
      float sc[8];
      #pragma unroll
      for (int r = 0; r < 8; ++r) {
        float mx = fmaxf(s[0][r], s[1][r]);
        #pragma unroll
        for (int d = 1; d < 16; d <<= 1) mx = fmaxf(mx, __shfl_xor(mx, d));
        const float nm = fmaxf(runmax[r], mx);
        sc[r] = __expf(runmax[r] - nm);
        const float p0 = __expf(s[0][r] - nm);
        const float p1 = __expf(s[1][r] - nm);
        s[0][r] = p0; s[1][r] = p1;
        float ts = p0 + p1;
        #pragma unroll
        for (int d = 1; d < 16; d <<= 1) ts += __shfl_xor(ts, d);
        runsum[r] = runsum[r] * sc[r] + ts;
        runmax[r] = nm;
      }
      // publish P (bf16, row-major 16x32) and row rescale factors
      #pragma unroll
      for (int r = 0; r < 8; ++r) {
        const int row = r + 8 * lg;
        pLds[row * 32 + li]      = (__bf16)s[0][r];
        pLds[row * 32 + 16 + li] = (__bf16)s[1][r];
      }
      if (li == 0)
        #pragma unroll
        for (int r = 0; r < 8; ++r) scaleLds[r + 8 * lg] = sc[r];
    }
    __syncthreads();

    // all waves: rescale O, then O += P(16x32) . V(32x16) per col-tile
    v16bf pf;
    {
      const __bf16* p = &pLds[li * 32 + lg * 16];
      pf = ldfrag(p, p + 8);
    }
    float scl[8];
    #pragma unroll
    for (int r = 0; r < 8; ++r) scl[r] = scaleLds[r + 8 * lg];
    #pragma unroll
    for (int t = 0; t < 4; ++t)
      #pragma unroll
      for (int r = 0; r < 8; ++r) o[t][r] *= scl[r];
    #pragma unroll
    for (int t = 0; t < 4; ++t) {
      const int c = cw + t * 16 + li;
      const __bf16* vp = Vb + (size_t)c * N_ + m0 + lg * 16;
      v16bf vf = ldfrag(vp, vp + 8);
      o[t] = wmma_bf16(pf, vf, o[t]);
    }
    __syncthreads();
  }

  if (wave == 0 && li == 0)
    #pragma unroll
    for (int r = 0; r < 8; ++r) rsumLds[r + 8 * lg] = runsum[r];
  __syncthreads();

  const float g = gamma[0];
  float inv[8];
  #pragma unroll
  for (int r = 0; r < 8; ++r) inv[r] = 1.0f / rsumLds[r + 8 * lg];

  const float* xb = x   + (size_t)b * C_ * N_;
  float*       ob = out + (size_t)b * C_ * N_;
  #pragma unroll
  for (int t = 0; t < 4; ++t)
    #pragma unroll
    for (int r = 0; r < 8; ++r) {
      const int c = cw + t * 16 + li;
      const int n = n0 + r + 8 * lg;
      const size_t idx = (size_t)c * N_ + n;
      ob[idx] = g * o[t][r] * inv[r] + xb[idx];
    }
}

// ---------------------------------------------------------------------------
extern "C" void kernel_launch(void* const* d_in, const int* in_sizes, int n_in,
                              void* d_out, int out_size, void* d_ws, size_t ws_size,
                              hipStream_t stream) {
  const float* x     = (const float*)d_in[0];
  const float* Wq    = (const float*)d_in[1];
  const float* Wk    = (const float*)d_in[2];
  const float* Wv    = (const float*)d_in[3];
  const float* gamma = (const float*)d_in[4];
  float* out = (float*)d_out;

  char* ws = (char*)d_ws;
  const size_t xT_bytes   = (size_t)B_ * N_ * C_ * 2;   // 37,748,736
  const size_t qkt_bytes  = (size_t)B_ * N_ * 128 * 2;  //  9,437,184
  const size_t v_bytes    = (size_t)B_ * C_ * N_ * 2;   // 37,748,736
  const size_t wqkt_bytes = (size_t)C_ * 128 * 2;
  __bf16* xT   = (__bf16*)(ws);
  __bf16* QKT  = (__bf16*)(ws + xT_bytes);
  __bf16* Vw   = (__bf16*)(ws + xT_bytes + qkt_bytes);
  __bf16* WqkT = (__bf16*)(ws + xT_bytes + qkt_bytes + v_bytes);
  __bf16* Wvb  = (__bf16*)(ws + xT_bytes + qkt_bytes + v_bytes + wqkt_bytes);

  prep_x<<<dim3(N_ / 32, C_ / 32, B_), dim3(32, 8), 0, stream>>>(x, xT);
  prep_w<<<(C_ * C_) / 256, 256, 0, stream>>>(Wq, Wk, Wv, WqkT, Wvb);
  qk_gemm<<<dim3(N_ / 128, 1, B_), 256, 0, stream>>>(xT, WqkT, QKT);
  v_gemm<<<dim3(N_ / 128, C_ / 128, B_), 256, 0, stream>>>(x, Wvb, Vw);
  attn_fused<<<dim3(N_ / 16, B_), 256, 0, stream>>>(QKT, Vw, x, gamma, out);
}
